// MultiHeadAttention_60189671686376
// MI455X (gfx1250) — compile-verified, matrix operand order checked by reading, not run on hardware
//
#include <hip/hip_runtime.h>
#include <hip/hip_bf16.h>
#include <stdint.h>

// ---------------- types ----------------
typedef __bf16 bf16;
typedef __attribute__((ext_vector_type(16))) __bf16 v16bf;
typedef __attribute__((ext_vector_type(8)))  __bf16 bf16x8;
typedef __attribute__((ext_vector_type(8)))  float  v8f;
typedef __attribute__((ext_vector_type(4)))  unsigned tdm4u;
typedef __attribute__((ext_vector_type(8)))  int      tdm8i;
typedef __attribute__((ext_vector_type(4)))  int      tdm4i;

union Frag { v16bf v; bf16x8 h[2]; };

#define BB 2
#define NN 4096
#define CC 768
#define HH 12
#define HD 64
#define D3 2304   // 3*C
#define MM 8192   // B*N

#if defined(__has_builtin)
#if __has_builtin(__builtin_amdgcn_tensor_load_to_lds) && __has_builtin(__builtin_amdgcn_s_wait_tensorcnt)
#define HAVE_TDM 1
#endif
#endif

static __device__ __forceinline__ bf16 f2bf(float f) {
    union { float f; uint32_t u; } a; a.f = f;
    uint32_t r = a.u + 0x7FFFu + ((a.u >> 16) & 1u);   // RNE
    union { uint16_t s; bf16 b; } o; o.s = (uint16_t)(r >> 16);
    return o.b;
}

static __device__ __forceinline__ bf16x8 ld8(const bf16* p) {
    return *(const bf16x8*)p;
}

static __device__ __forceinline__ v8f wmma_bf16(const Frag& a, const Frag& b, v8f c) {
    return __builtin_amdgcn_wmma_f32_16x16x32_bf16(false, a.v, false, b.v,
                                                   (short)0, c, false, false);
}

#ifdef HAVE_TDM
// TDM 2-D tile load: global (row-major, stride0 elems/row) -> LDS, with per-row
// LDS padding inserted by the DMA engine (pad fields), bf16 elements (data_size=1).
// Descriptor encoding per CDNA5 ISA.
// This toolchain exposes the 6-arg builtin: (g0 v4u, g1 v8i, g2 v4i, g3 v4i, v8i, cpol).
static __device__ __forceinline__ void tdm_load_2d(uint32_t lds_off, const void* gptr,
                                                   uint32_t tile_d0, uint32_t tile_d1,
                                                   uint32_t tensor_d0, uint32_t tensor_d1,
                                                   uint32_t stride0,
                                                   uint32_t pad_interval, uint32_t pad_amount) {
    const uint64_t ga = (uint64_t)(uintptr_t)gptr;
    tdm4u g0;
    g0[0] = 1u;                                              // count=1, user mode
    g0[1] = lds_off;                                         // lds_addr (bytes)
    g0[2] = (uint32_t)ga;                                    // global_addr[31:0]
    g0[3] = (uint32_t)((ga >> 32) & 0x01FFFFFFu) | (2u << 30); // addr[56:32], type=2
    tdm8i g1;
    g1[0] = (int)((1u << 16) | (1u << 20) |                  // data_size=2B, pad_enable
                  (pad_interval << 22) | (pad_amount << 25));
    g1[1] = (int)((tensor_d0 & 0xFFFFu) << 16);              // atomic_addr=0 | tdim0.lo
    g1[2] = (int)((tensor_d0 >> 16) | ((tensor_d1 & 0xFFFFu) << 16));
    g1[3] = (int)((tensor_d1 >> 16) | (tile_d0 << 16));
    g1[4] = (int)tile_d1;                                    // tile_dim2 = 0 (2-D)
    g1[5] = (int)stride0;                                    // dim0 stride, elems
    g1[6] = 0;
    g1[7] = 0;
    const tdm4i z4 = (tdm4i){0, 0, 0, 0};
    const tdm8i z8 = (tdm8i){0, 0, 0, 0, 0, 0, 0, 0};
    __builtin_amdgcn_tensor_load_to_lds(g0, g1, z4, z4, z8, 0);
}
#endif

// ---------------- kernel 1: f32 -> bf16 ----------------
__global__ void cvt_f32_bf16(const float* __restrict__ in, bf16* __restrict__ out, int n) {
    int i = blockIdx.x * 256 + threadIdx.x;
    if (i < n) out[i] = f2bf(in[i]);
}

// ---------------- kernel 2: generic bf16 GEMM  C[m,n] = sum_k A[m,k]*Bw[n,k] + bias[n]
__global__ __launch_bounds__(128)
void gemm_bf16(const bf16* __restrict__ A, const bf16* __restrict__ Bw,
               const float* __restrict__ bias, float* __restrict__ Cc,
               int K, int ldc) {
    const int lane  = threadIdx.x & 31;
    const int w     = threadIdx.x >> 5;
    const int l16   = lane & 15;
    const int half8 = (lane >> 4) * 8;
    const int mbase = blockIdx.x * 64  + (w >> 1) * 32;
    const int nbase = blockIdx.y * 128 + (w & 1) * 64;

    v8f acc[2][4];
#pragma unroll
    for (int mc = 0; mc < 2; ++mc)
#pragma unroll
        for (int nc = 0; nc < 4; ++nc) acc[mc][nc] = (v8f){0,0,0,0,0,0,0,0};

    for (int k0 = 0; k0 < K; k0 += 32) {
        Frag af[2], bfr[4];
#pragma unroll
        for (int mc = 0; mc < 2; ++mc) {
            const bf16* p = A + (size_t)(mbase + mc*16 + l16) * K + k0;
            af[mc].h[0] = ld8(p + half8);
            af[mc].h[1] = ld8(p + 16 + half8);
            __builtin_prefetch(p + 128, 0, 3);  // locality 3 -> WGP scope: all cache levels
        }
#pragma unroll
        for (int nc = 0; nc < 4; ++nc) {
            const bf16* p = Bw + (size_t)(nbase + nc*16 + l16) * K + k0;
            bfr[nc].h[0] = ld8(p + half8);
            bfr[nc].h[1] = ld8(p + 16 + half8);
            __builtin_prefetch(p + 128, 0, 3);
        }
#pragma unroll
        for (int mc = 0; mc < 2; ++mc)
#pragma unroll
            for (int nc = 0; nc < 4; ++nc)
                acc[mc][nc] = wmma_bf16(af[mc], bfr[nc], acc[mc][nc]);
    }

#pragma unroll
    for (int nc = 0; nc < 4; ++nc) {
        const float bs = bias[nbase + nc*16 + l16];
#pragma unroll
        for (int mc = 0; mc < 2; ++mc) {
            const int mrow = mbase + mc*16 + (lane >> 4) * 8;
#pragma unroll
            for (int r = 0; r < 8; ++r)
                Cc[(size_t)(mrow + r) * ldc + nbase + nc*16 + l16] = acc[mc][nc][r] + bs;
        }
    }
}

// ---------------- kernel 3: RoPE + scatter ----------------
__global__ void rope_scatter(const float* __restrict__ qkv,
                             bf16* __restrict__ Qr, bf16* __restrict__ Kr,
                             bf16* __restrict__ Vt) {
    int idx = blockIdx.x * 256 + threadIdx.x;          // over B*N*H*32
    const int e = idx & 31;
    idx >>= 5;
    const int h = idx % HH;  idx /= HH;
    const int n = idx % NN;
    const int b = idx / NN;

    const size_t base = (size_t)(b * NN + n) * D3 + h * HD + e;
    const float q1 = qkv[base];              const float q2 = qkv[base + 32];
    const float k1 = qkv[base + CC];         const float k2 = qkv[base + CC + 32];
    const float v1 = qkv[base + 2*CC];       const float v2 = qkv[base + 2*CC + 32];

    const float inv = __expf(-(float)(2 * e) / (float)HD * 9.210340371976184f); // ln(1e4)
    const float fr  = (float)n * inv;
    const float c = __cosf(fr), s = __sinf(fr);

    const float scale = 0.125f;  // HD^-0.5 folded into Q
    const float qo1 = (q1 * c - q2 * s) * scale;
    const float qo2 = (q2 * c + q1 * s) * scale;
    const float ko1 =  k1 * c - k2 * s;
    const float ko2 =  k2 * c + k1 * s;

    const size_t ro = ((size_t)(b * HH + h) * NN + n) * HD + e;
    Qr[ro]      = f2bf(qo1);  Qr[ro + 32]      = f2bf(qo2);
    Kr[ro]      = f2bf(ko1);  Kr[ro + 32]      = f2bf(ko2);
    const size_t to = ((size_t)(b * HH + h) * HD + e) * NN + n;
    Vt[to] = f2bf(v1);  Vt[to + (size_t)32 * NN] = f2bf(v2);
}

// ---------------- kernel 4: flash attention ----------------
// One block = one (b,h) x 128 query rows; 8 waves, wave w owns q-rows [base+16w, base+16w+16).
// K/V tiles (32 keys) staged to LDS by the Tensor Data Mover, double-buffered:
// DMA of step s+1 overlaps WMMA of step s.  S^T = K x Q^T so P^T (C layout) feeds the
// O^T WMMA's B operand with zero cross-lane movement;  O^T = V^T x P^T.
#define KROW 72   // 64 bf16 data + 8 pad  (pad_interval=4: 128B, pad_amount=3: 16B)
#define VROW 40   // 32 bf16 data + 8 pad  (pad_interval=3:  64B, pad_amount=3: 16B)
#define NSTEP (NN / 32)

__global__ __launch_bounds__(256)
void flash_attn(const bf16* __restrict__ Qr, const bf16* __restrict__ Kr,
                const bf16* __restrict__ Vt, bf16* __restrict__ Oa) {
    __shared__ __align__(16) bf16 sK[2][32 * KROW];
    __shared__ __align__(16) bf16 sV[2][64 * VROW];

    const int lane  = threadIdx.x & 31;
    const int w     = threadIdx.x >> 5;
    const int l16   = lane & 15;
    const int half8 = (lane >> 4) * 8;
    const int bh    = blockIdx.x >> 5;          // 0..23
    const int qtile = (blockIdx.x & 31) * 8 + w;

    // Q^T B-fragments (col = q), loaded once from global
    Frag qf[2];
    {
        const bf16* qb = Qr + ((size_t)bh * NN + qtile * 16 + l16) * HD;
#pragma unroll
        for (int c = 0; c < 2; ++c) {
            qf[c].h[0] = ld8(qb + c*32 + half8);
            qf[c].h[1] = ld8(qb + c*32 + 16 + half8);
        }
    }

    const bf16* Kbase = Kr + (size_t)bh * NN * HD;
    const bf16* Vbase = Vt + (size_t)bh * HD * NN;

    v8f o[4];
#pragma unroll
    for (int nn = 0; nn < 4; ++nn) o[nn] = (v8f){0,0,0,0,0,0,0,0};
    float m = -1e30f, lsum = 0.0f;

#ifdef HAVE_TDM
    if (threadIdx.x < 32) {   // wave 0 drives the TDM (EXEC ignored; one issue per WG)
        tdm_load_2d((uint32_t)(uintptr_t)&sK[0][0], Kbase, HD, 32, HD, NN, HD, 4, 3);
        tdm_load_2d((uint32_t)(uintptr_t)&sV[0][0], Vbase, 32, HD, NN, HD, NN, 3, 3);
    }
#endif

    for (int s = 0; s < NSTEP; ++s) {
        const int buf = s & 1;
        const int j2  = s * 32;

#ifdef HAVE_TDM
        if (threadIdx.x < 32) {
            if (s + 1 < NSTEP) {
                tdm_load_2d((uint32_t)(uintptr_t)&sK[buf ^ 1][0], Kbase + (size_t)(j2 + 32) * HD,
                            HD, 32, HD, NN, HD, 4, 3);
                tdm_load_2d((uint32_t)(uintptr_t)&sV[buf ^ 1][0], Vbase + (j2 + 32),
                            32, HD, NN, HD, NN, 3, 3);
                __builtin_amdgcn_s_wait_tensorcnt(2);   // pair for step s complete
            } else {
                __builtin_amdgcn_s_wait_tensorcnt(0);
            }
        }
        __syncthreads();
#else
        {   // fallback: cooperative b128 staging
            const int t = threadIdx.x;
            const int kr = t >> 3, kc = t & 7;     // 32 rows x 8 chunks of 16B
            *(bf16x8*)&sK[buf][kr * KROW + kc * 8] = ld8(Kbase + (size_t)(j2 + kr) * HD + kc * 8);
            const int vr = t >> 2, vc = t & 3;     // 64 rows x 4 chunks of 16B
            *(bf16x8*)&sV[buf][vr * VROW + vc * 8] = ld8(Vbase + (size_t)vr * NN + j2 + vc * 8);
        }
        __syncthreads();
#endif

        // --- S^T tiles from LDS K: s[t][r] = S^T[key j2+t*16+r+8*(lane/16)][q l16]
        v8f s0 = (v8f){0,0,0,0,0,0,0,0};
        v8f s1 = (v8f){0,0,0,0,0,0,0,0};
#pragma unroll
        for (int t = 0; t < 2; ++t) {
            const bf16* kb = &sK[buf][(t * 16 + l16) * KROW];
            Frag af;
#pragma unroll
            for (int c = 0; c < 2; ++c) {
                af.h[0] = *(const bf16x8*)(kb + c*32 + half8);
                af.h[1] = *(const bf16x8*)(kb + c*32 + 16 + half8);
                if (t == 0) s0 = wmma_bf16(af, qf[c], s0);
                else        s1 = wmma_bf16(af, qf[c], s1);
            }
        }

        // --- online softmax (per q column = per lane; lanes l and l^16 share q)
        float mstep = s0[0];
#pragma unroll
        for (int r = 0; r < 8; ++r) { mstep = fmaxf(mstep, s0[r]); mstep = fmaxf(mstep, s1[r]); }
        mstep = fmaxf(mstep, __shfl_xor(mstep, 16, 32));
        const float mnew  = fmaxf(m, mstep);
        const float alpha = __expf(m - mnew);

        float p0[8], p1[8], loc = 0.0f;
#pragma unroll
        for (int r = 0; r < 8; ++r) {
            p0[r] = __expf(s0[r] - mnew);
            p1[r] = __expf(s1[r] - mnew);
            loc += p0[r] + p1[r];
        }
        loc += __shfl_xor(loc, 16, 32);
        lsum = lsum * alpha + loc;
        m = mnew;
#pragma unroll
        for (int nn = 0; nn < 4; ++nn)
#pragma unroll
            for (int r = 0; r < 8; ++r) o[nn][r] *= alpha;

        // --- P^T as B fragment: element i<8 -> s0[i], i>=8 -> s1[i-8] (same lane)
        Frag pf;
#pragma unroll
        for (int i = 0; i < 8; ++i) { pf.v[i] = f2bf(p0[i]); pf.v[i + 8] = f2bf(p1[i]); }

        // --- O^T += V^T x P^T  (V^T tiles from LDS)
#pragma unroll
        for (int nn = 0; nn < 4; ++nn) {
            const bf16* vb = &sV[buf][(nn * 16 + l16) * VROW];
            Frag af;
            af.h[0] = *(const bf16x8*)(vb + half8);
            af.h[1] = *(const bf16x8*)(vb + 16 + half8);
            o[nn] = wmma_bf16(af, pf, o[nn]);
        }

        __syncthreads();   // all waves done with buf before DMA of step s+1 overwrites it
    }

    const float invl = 1.0f / lsum;
    const int b = bh / HH, h = bh % HH;
    const int q = qtile * 16 + l16;
#pragma unroll
    for (int nn = 0; nn < 4; ++nn)
#pragma unroll
        for (int r = 0; r < 8; ++r) {
            const int hd = nn*16 + r + 8*(lane >> 4);
            Oa[((size_t)b * NN + q) * CC + h * HD + hd] = f2bf(o[nn][r] * invl);
        }
}

// ---------------- launcher ----------------
extern "C" void kernel_launch(void* const* d_in, const int* in_sizes, int n_in,
                              void* d_out, int out_size, void* d_ws, size_t ws_size,
                              hipStream_t stream) {
    const float* x      = (const float*)d_in[0];
    const float* w_qkv  = (const float*)d_in[1];
    const float* b_qkv  = (const float*)d_in[2];
    const float* w_proj = (const float*)d_in[3];
    const float* b_proj = (const float*)d_in[4];
    float* out = (float*)d_out;

    char* ws = (char*)d_ws;
    bf16*  xb   = (bf16*)ws;   ws += (size_t)MM * CC * 2;            // 12.6 MB
    bf16*  wqb  = (bf16*)ws;   ws += (size_t)D3 * CC * 2;            //  3.5 MB
    bf16*  wpb  = (bf16*)ws;   ws += (size_t)CC * CC * 2;            //  1.2 MB
    float* qkvf = (float*)ws;  ws += (size_t)MM * D3 * 4;            // 75.5 MB
    bf16*  Qr   = (bf16*)ws;   ws += (size_t)BB * HH * NN * HD * 2;  // 12.6 MB
    bf16*  Kr   = (bf16*)ws;   ws += (size_t)BB * HH * NN * HD * 2;
    bf16*  Vt   = (bf16*)ws;   ws += (size_t)BB * HH * NN * HD * 2;
    bf16*  attn = (bf16*)qkvf;  // alias: qkv f32 is dead after rope_scatter

    int n = MM * CC;
    cvt_f32_bf16<<<(n + 255) / 256, 256, 0, stream>>>(x, xb, n);
    n = D3 * CC;
    cvt_f32_bf16<<<(n + 255) / 256, 256, 0, stream>>>(w_qkv, wqb, n);
    n = CC * CC;
    cvt_f32_bf16<<<(n + 255) / 256, 256, 0, stream>>>(w_proj, wpb, n);

    // QKV GEMM: [8192,768] x [2304,768]^T -> [8192,2304] f32 (+bias)
    gemm_bf16<<<dim3(MM / 64, D3 / 128), 128, 0, stream>>>(xb, wqb, b_qkv, qkvf, CC, D3);

    // RoPE + scatter (Q scaled), V transposed
    n = BB * NN * HH * 32;
    rope_scatter<<<n / 256, 256, 0, stream>>>(qkvf, Qr, Kr, Vt);

    // attention: 24 heads x 32 q-blocks (128 rows each), 8 waves/block
    flash_attn<<<BB * HH * (NN / 128), 256, 0, stream>>>(Qr, Kr, Vt, attn);

    // projection: [8192,768] x [768,768]^T -> d_out (+bias)
    gemm_bf16<<<dim3(MM / 64, CC / 128), 128, 0, stream>>>(attn, wpb, b_proj, out, CC, CC);
}
